// RQ_6614249636302
// MI455X (gfx1250) — matrix operand order repaired and checked by reading, not compiled, not run on hardware
//
#include <hip/hip_runtime.h>
#include <stdint.h>

typedef __attribute__((ext_vector_type(16))) __bf16 v16bf;
typedef __attribute__((ext_vector_type(8)))  float  v8f;

#define DIM   128
#define MLEV  4
#define KCB   256
#define ROWS_PER_BLOCK 128
#define LDS_ROW_BYTES 272           // 256B bf16 row + 16B pad (68 dwords: shifts banks by 4/row)
#define HALF_CODES 128              // stage 128 codes (32KB payload) at a time

union BfVec { v16bf v; uint32_t u[8]; __bf16 b[16]; };

__device__ __forceinline__ void load8(const float* __restrict__ p, float* o) {
    const float4* q = (const float4*)p;
    float4 x = q[0], y = q[1];
    o[0]=x.x; o[1]=x.y; o[2]=x.z; o[3]=x.w;
    o[4]=y.x; o[5]=y.y; o[6]=y.z; o[7]=y.w;
}
__device__ __forceinline__ void store8(float* __restrict__ p, const float* o) {
    float4 x = {o[0],o[1],o[2],o[3]};
    float4 y = {o[4],o[5],o[6],o[7]};
    float4* q = (float4*)p;
    q[0]=x; q[1]=y;
}

// ---- Prep 1: codebooks f32 -> bf16 (4 elements / thread) ----
__global__ __launch_bounds__(256)
void cvt_bf16_kernel(const float* __restrict__ cb, unsigned short* __restrict__ out) {
    const int i = (blockIdx.x * 256 + threadIdx.x) * 4;
    float4 v = *(const float4*)(cb + i);
    __bf16* o = (__bf16*)(out + i);
    o[0] = (__bf16)v.x; o[1] = (__bf16)v.y; o[2] = (__bf16)v.z; o[3] = (__bf16)v.w;
}

// ---- Prep 2: csq[m*K+k] = sum_d cb[m][k][d]^2  (one code / thread) ----
__global__ __launch_bounds__(256)
void csq_kernel(const float* __restrict__ cb, float* __restrict__ csq) {
    const int code = blockIdx.x * 256 + threadIdx.x;
    const float* row = cb + (size_t)code * DIM;
    float s = 0.f;
    #pragma unroll
    for (int d = 0; d < DIM; d += 4) {
        float4 v = *(const float4*)(row + d);
        s += v.x*v.x + v.y*v.y + v.z*v.z + v.w*v.w;
    }
    csq[code] = s;
}

// ---- Main fused RQ kernel ----
__global__ __launch_bounds__(256)
void rq_wmma_kernel(const float* __restrict__ data,
                    const float* __restrict__ codebooks,
                    const unsigned short* __restrict__ cb_bf16,
                    const float* __restrict__ csq_all,
                    float* __restrict__ out_vq,
                    float* __restrict__ out_ids)
{
    __shared__ __align__(16) unsigned char ldsCB[HALF_CODES * LDS_ROW_BYTES]; // 34816 B
    __shared__ int ldsIdx[8 * 16];

    const int lane = threadIdx.x & 31;
    const int wave = threadIdx.x >> 5;
    const int h    = lane >> 4;          // half-wave selector (K sub-range)
    const int col  = lane & 15;          // A-row / B-col / D-col index
    const int rBase = blockIdx.x * ROWS_PER_BLOCK + wave * 16;
    const int myRow = rBase + col;

    // staging decomposition: thread -> fixed 16B chunk column, row advances by 16
    const int sRow   = threadIdx.x >> 4;   // 0..15
    const int sChunk = threadIdx.x & 15;   // 16B chunk within a 256B row

    const float* __restrict__ drow = data + (size_t)myRow * DIM;

    // Residual in registers, laid out exactly as the WMMA bf16 A-matrix:
    float res[64];
    #pragma unroll
    for (int c = 0; c < 4; ++c) {
        load8(drow + c*32 + h*8,      &res[c*16]);
        load8(drow + c*32 + 16 + h*8, &res[c*16 + 8]);
    }

    #pragma unroll 1
    for (int lev = 0; lev < MLEV; ++lev) {
        const float* __restrict__ cbf  = codebooks + (size_t)lev * KCB * DIM;
        const float* __restrict__ csqL = csq_all   + (size_t)lev * KCB;
        const unsigned char* cbbLev = (const unsigned char*)(cb_bf16 + (size_t)lev * KCB * DIM);

        // Residual -> bf16 A (v_cvt_pk_bf16_f32 via native casts)
        BfVec A[4];
        #pragma unroll
        for (int c = 0; c < 4; ++c) {
            #pragma unroll
            for (int e = 0; e < 16; ++e)
                A[c].b[e] = (__bf16)res[c*16 + e];
        }

        float    minv[8];
        uint32_t mini[8];
        #pragma unroll
        for (int v = 0; v < 8; ++v) { minv[v] = 3.4e38f; mini[v] = 0u; }

        #pragma unroll 1
        for (int half = 0; half < 2; ++half) {
            // ---- Async-stage 128 codes (32KB) of bf16 codebook into padded LDS ----
            #pragma unroll
            for (int i = 0; i < 8; ++i) {
                const int r = i * 16 + sRow;                               // 0..127
                uint32_t ldsA = (uint32_t)(uintptr_t)(ldsCB + r * LDS_ROW_BYTES + sChunk * 16);
                uint64_t ga   = (uint64_t)(uintptr_t)(cbbLev
                                + ((size_t)(half * HALF_CODES + r) * 256 + sChunk * 16));
                asm volatile("global_load_async_to_lds_b128 %0, %1, off"
                             :: "v"(ldsA), "v"(ga) : "memory");
            }
            asm volatile("s_wait_asynccnt 0x0" ::: "memory");
            __syncthreads();

            // ---- 8 column tiles against the staged half ----
            #pragma unroll 1
            for (int t = 0; t < HALF_CODES / 16; ++t) {
                const int rloc = t * 16 + col;                 // local code row in LDS
                const int code = half * HALF_CODES + rloc;     // global code id

                BfVec B[4];
                #pragma unroll
                for (int c = 0; c < 4; ++c) {
                    const uint4* p = (const uint4*)&ldsCB[rloc * LDS_ROW_BYTES + c*64 + h*16];
                    uint4 x = p[0], y = p[2];                  // ds_load_b128 x2
                    B[c].u[0]=x.x; B[c].u[1]=x.y; B[c].u[2]=x.z; B[c].u[3]=x.w;
                    B[c].u[4]=y.x; B[c].u[5]=y.y; B[c].u[6]=y.z; B[c].u[7]=y.w;
                }
                const float csq = csqL[code];                  // exact f32 ||c||^2

                v8f acc = {0.f,0.f,0.f,0.f,0.f,0.f,0.f,0.f};
                acc = __builtin_amdgcn_wmma_f32_16x16x32_bf16(false, A[0].v, false, B[0].v, (short)0, acc, false, false);
                acc = __builtin_amdgcn_wmma_f32_16x16x32_bf16(false, A[1].v, false, B[1].v, (short)0, acc, false, false);
                acc = __builtin_amdgcn_wmma_f32_16x16x32_bf16(false, A[2].v, false, B[2].v, (short)0, acc, false, false);
                acc = __builtin_amdgcn_wmma_f32_16x16x32_bf16(false, A[3].v, false, B[3].v, (short)0, acc, false, false);

                #pragma unroll
                for (int v = 0; v < 8; ++v) {
                    float s = fmaf(-2.0f, acc[v], csq);        // ||c||^2 - 2 res.c
                    if (s < minv[v]) { minv[v] = s; mini[v] = (uint32_t)code; }
                }
            }
            __syncthreads();   // done reading this half before restaging
        }

        // Min+argmin across the 16 lanes sharing each D-row (xor 1,2,4,8)
        #pragma unroll
        for (int off = 1; off <= 8; off <<= 1) {
            #pragma unroll
            for (int v = 0; v < 8; ++v) {
                float    ov = __shfl_xor(minv[v], off);
                uint32_t oi = (uint32_t)__shfl_xor((int)mini[v], off);
                if (ov < minv[v] || (ov == minv[v] && oi < mini[v])) {
                    minv[v] = ov; mini[v] = oi;
                }
            }
        }

        if (col == 0) {
            #pragma unroll
            for (int v = 0; v < 8; ++v)
                ldsIdx[wave*16 + h*8 + v] = (int)mini[v];      // D row m = v + 8h
        }
        __syncthreads();

        if (lane < 16)
            out_ids[(size_t)(rBase + lane) * MLEV + lev] = (float)ldsIdx[wave*16 + lane];

        // Gather selected codeword in exact f32, update residual in-place
        const int codeSel = ldsIdx[wave*16 + col];
        const float* __restrict__ erow = cbf + (size_t)codeSel * DIM;
        #pragma unroll
        for (int c = 0; c < 4; ++c) {
            float e0[8], e1[8];
            load8(erow + c*32 + h*8,      e0);
            load8(erow + c*32 + 16 + h*8, e1);
            #pragma unroll
            for (int i = 0; i < 8; ++i) { res[c*16+i] -= e0[i]; res[c*16+8+i] -= e1[i]; }
        }
        __syncthreads();   // protect ldsIdx before the next level rewrites it
    }

    // vq_emb = data - final_residual
    float* __restrict__ orow = out_vq + (size_t)myRow * DIM;
    #pragma unroll
    for (int c = 0; c < 4; ++c) {
        float t0[8], t1[8];
        load8(drow + c*32 + h*8,      t0);
        load8(drow + c*32 + 16 + h*8, t1);
        #pragma unroll
        for (int i = 0; i < 8; ++i) { t0[i] -= res[c*16+i]; t1[i] -= res[c*16+8+i]; }
        store8(orow + c*32 + h*8,      t0);
        store8(orow + c*32 + 16 + h*8, t1);
    }
}

extern "C" void kernel_launch(void* const* d_in, const int* in_sizes, int n_in,
                              void* d_out, int out_size, void* d_ws, size_t ws_size,
                              hipStream_t stream) {
    const float* data = (const float*)d_in[0];          // [N, 128] f32
    const float* cbs  = (const float*)d_in[1];          // [4, 256, 128] f32
    const int n = in_sizes[0] / DIM;

    float* out_vq  = (float*)d_out;                     // [N, 128] f32
    float* out_ids = out_vq + (size_t)n * DIM;          // [N, 4] as f32

    // Workspace: bf16 codebook (256 KB) then csq (4 KB)
    unsigned short* cb_bf16 = (unsigned short*)d_ws;
    float*          csq     = (float*)((char*)d_ws + (size_t)MLEV * KCB * DIM * sizeof(unsigned short));

    const int cvt_elems = MLEV * KCB * DIM;             // 131072
    cvt_bf16_kernel<<<dim3(cvt_elems / (256 * 4)), dim3(256), 0, stream>>>(cbs, cb_bf16);
    csq_kernel<<<dim3((MLEV * KCB) / 256), dim3(256), 0, stream>>>(cbs, csq);

    rq_wmma_kernel<<<dim3(n / ROWS_PER_BLOCK), dim3(256), 0, stream>>>(
        data, cbs, cb_bf16, csq, out_vq, out_ids);
}
